// GNN3_61924838474116
// MI455X (gfx1250) — matrix operand order repaired and checked by reading, not compiled, not run on hardware
//
#include <hip/hip_runtime.h>
#include <hip/hip_bf16.h>
#include <math.h>
#include <string.h>

#define G_    128
#define NPER  64
#define NN    8192
#define EE    65536
#define FEATN 30
#define EFEAT 11
#define EMB   256
#define HT    4
#define HG    3

typedef __bf16 bf16_t;
typedef bf16_t bf16x8  __attribute__((ext_vector_type(8)));
typedef bf16_t bf16x16 __attribute__((ext_vector_type(16)));
typedef float  f32x8   __attribute__((ext_vector_type(8)));

__device__ __forceinline__ bf16_t f2bf(float f) {
  unsigned u = __builtin_bit_cast(unsigned, f);
  unsigned r = u + 0x7FFFu + ((u >> 16) & 1u);   // RNE
  unsigned short h = (unsigned short)(r >> 16);
  return __builtin_bit_cast(bf16_t, h);
}
__device__ __forceinline__ unsigned fenc(float f) {
  unsigned u = __builtin_bit_cast(unsigned, f);
  return (u & 0x80000000u) ? ~u : (u | 0x80000000u);
}
__device__ __forceinline__ float fdec(unsigned u) {
  unsigned v = (u & 0x80000000u) ? (u & 0x7FFFFFFFu) : ~u;
  return __builtin_bit_cast(float, v);
}

// ---------------------------------------------------------------- WMMA GEMM
// C[M,Nc] = act( A[M,K](bf16) * BT[Nc,K](bf16)^T + bias ), optional C-accumulate.
// Register-blocked: each wave computes a 32x64 tile = 2x4 grid of 16x16 WMMA
// tiles -> 8 v_wmma_f32_16x16x32_bf16 per 12 b128 loads per K-step.
#define GEMM_WAVES 4
#define TM 2
#define TN 4
__global__ __launch_bounds__(32 * GEMM_WAVES)
void k_gemm_bf16(const bf16_t* __restrict__ A, const bf16_t* __restrict__ BT,
                 float* __restrict__ C, const float* __restrict__ bias,
                 int M, int Nc, int K, int relu, int accum) {
  const int wave = threadIdx.x >> 5;
  const int lane = threadIdx.x & 31;
  const int half = lane >> 4;
  const int l16  = lane & 15;
  const int n0 = (blockIdx.x * GEMM_WAVES + wave) * (16 * TN);
  const int m0 = blockIdx.y * (16 * TM);
  if (n0 >= Nc || m0 >= M) return;                 // wave-uniform
  const bf16_t* __restrict__ arow[TM];
  const bf16_t* __restrict__ brow[TN];
#pragma unroll
  for (int t = 0; t < TM; ++t) arow[t] = A  + (size_t)(m0 + t * 16 + l16) * K;
#pragma unroll
  for (int t = 0; t < TN; ++t) brow[t] = BT + (size_t)(n0 + t * 16 + l16) * K;

  f32x8 cfrag[TM][TN];
#pragma unroll
  for (int tm = 0; tm < TM; ++tm)
#pragma unroll
    for (int tn = 0; tn < TN; ++tn) cfrag[tm][tn] = (f32x8){};

  for (int k0 = 0; k0 < K; k0 += 32) {
    bf16x16 afrag[TM], bfrag[TN];
#pragma unroll
    for (int t = 0; t < TM; ++t) {
      // A 16x32 bf16 lane layout: lanes0-15 K{0..7,16..23}, lanes16-31 K{8..15,24..31}
      bf16x8 a_lo = *(const bf16x8*)(arow[t] + k0 + half * 8);
      bf16x8 a_hi = *(const bf16x8*)(arow[t] + k0 + 16 + half * 8);
      afrag[t] = __builtin_shufflevector(a_lo, a_hi, 0,1,2,3,4,5,6,7,8,9,10,11,12,13,14,15);
      __builtin_prefetch(arow[t] + k0 + 256, 0, 3);   // global_prefetch_b8, near-cache
    }
#pragma unroll
    for (int t = 0; t < TN; ++t) {
      // B 32x16 bf16: lane (n) holds K{0..7,16..23} (lanes0-15) or K{8..15,24..31} (lanes16-31)
      bf16x8 b_lo = *(const bf16x8*)(brow[t] + k0 + half * 8);
      bf16x8 b_hi = *(const bf16x8*)(brow[t] + k0 + 16 + half * 8);
      bfrag[t] = __builtin_shufflevector(b_lo, b_hi, 0,1,2,3,4,5,6,7,8,9,10,11,12,13,14,15);
      __builtin_prefetch(brow[t] + k0 + 256, 0, 3);
    }
#pragma unroll
    for (int tm = 0; tm < TM; ++tm)
#pragma unroll
      for (int tn = 0; tn < TN; ++tn)
        cfrag[tm][tn] = __builtin_amdgcn_wmma_f32_16x16x32_bf16(
            false, afrag[tm], false, bfrag[tn], (short)0, cfrag[tm][tn], false, false);
  }

#pragma unroll
  for (int tn = 0; tn < TN; ++tn) {
    const int col = n0 + tn * 16 + l16;
    const float bv = bias ? bias[col] : 0.0f;
#pragma unroll
    for (int tm = 0; tm < TM; ++tm) {
      const int rowbase = m0 + tm * 16 + half * 8;   // C layout: VGPR r -> M = r (+8 hi half)
#pragma unroll
      for (int r = 0; r < 8; ++r) {
        size_t o = (size_t)(rowbase + r) * Nc + col;
        float v = cfrag[tm][tn][r] + bv;
        if (accum) v += C[o];
        if (relu)  v = fmaxf(v, 0.0f);
        C[o] = v;
      }
    }
  }
}

// ---------------------------------------------------------------- utility
__global__ void k_zero_f32(float* p, int n) { int i = blockIdx.x*blockDim.x+threadIdx.x; if (i < n) p[i] = 0.f; }
__global__ void k_fill_int(int* p, int v, int n) { int i = blockIdx.x*blockDim.x+threadIdx.x; if (i < n) p[i] = v; }
__global__ void k_fill_u32(unsigned* p, unsigned v, int n) { int i = blockIdx.x*blockDim.x+threadIdx.x; if (i < n) p[i] = v; }
__global__ void k_copy_f32(const float* a, float* b, int n) { int i = blockIdx.x*blockDim.x+threadIdx.x; if (i < n) b[i] = a[i]; }
__global__ void k_cvt_bf16(const float* __restrict__ a, bf16_t* __restrict__ b, int n) {
  int i = blockIdx.x*blockDim.x+threadIdx.x; if (i < n) b[i] = f2bf(a[i]);
}
__global__ void k_cvt_pad(const float* __restrict__ a, bf16_t* __restrict__ b, int M, int K, int Kp) {
  int i = blockIdx.x*blockDim.x+threadIdx.x; if (i >= M * Kp) return;
  int m = i / Kp, k = i % Kp;
  b[i] = (k < K) ? f2bf(a[(size_t)m * K + k]) : f2bf(0.f);
}
// W[K,Nc] (f32) -> WT[Nc,Kp] (bf16, transposed, K zero-padded to Kp)
__global__ void k_wt(const float* __restrict__ w, bf16_t* __restrict__ wt, int K, int Nc, int Kp) {
  int i = blockIdx.x*blockDim.x+threadIdx.x; if (i >= Nc * Kp) return;
  int n = i / Kp, k = i % Kp;
  wt[i] = (k < K) ? f2bf(w[(size_t)k * Nc + n]) : f2bf(0.f);
}

// ---------------------------------------------------------------- GIN
__global__ void k_gin_scatter(const float* __restrict__ x, const int* __restrict__ src,
                              const int* __restrict__ dst, float* h) {
  int i = blockIdx.x*blockDim.x+threadIdx.x; if (i >= EE * FEATN) return;
  int e = i / FEATN, f = i % FEATN;
  atomicAdd(&h[(size_t)dst[e] * FEATN + f], x[(size_t)src[e] * FEATN + f]);
}

// ---------------------------------------------------------------- attention
__global__ void k_attn_logits(const float* __restrict__ q, const float* __restrict__ k,
                              const int* __restrict__ src, const int* __restrict__ dst,
                              float* __restrict__ lg) {
  int i = blockIdx.x*blockDim.x+threadIdx.x; if (i >= EE * HT) return;
  int e = i / HT, h = i % HT;
  const float4* qp = (const float4*)(q + (size_t)dst[e] * (HT*EMB) + h * EMB);
  const float4* kp = (const float4*)(k + (size_t)src[e] * (HT*EMB) + h * EMB);
  float s = 0.f;
  for (int j = 0; j < EMB/4; ++j) { float4 a = qp[j], b = kp[j]; s += a.x*b.x + a.y*b.y + a.z*b.z + a.w*b.w; }
  lg[i] = s * 0.0625f;   // 1/sqrt(256)
}
__global__ void k_seg_max(const float* lg, const int* dstv, const int* mask,
                          unsigned* m, int ne, int H) {
  int i = blockIdx.x*blockDim.x+threadIdx.x; if (i >= ne * H) return;
  int e = i / H, h = i % H;
  if (mask && !mask[e]) return;
  atomicMax(&m[dstv[e] * H + h], fenc(lg[i]));
}
__global__ void k_seg_exp(const float* lg, const int* dstv, const int* mask,
                          const unsigned* m, float* ev, float* den, int ne, int H) {
  int i = blockIdx.x*blockDim.x+threadIdx.x; if (i >= ne * H) return;
  int e = i / H, h = i % H;
  float out = 0.f;
  if (!mask || mask[e]) {
    float mm = fdec(m[dstv[e] * H + h]);
    if (!(mm > -1e29f)) mm = 0.f;
    out = __expf(lg[i] - mm);
    atomicAdd(&den[dstv[e] * H + h], out);
  }
  ev[i] = out;
}
__global__ void k_seg_div(float* ev, const float* den, const int* dstv, int ne, int H) {
  int i = blockIdx.x*blockDim.x+threadIdx.x; if (i >= ne * H) return;
  int e = i / H, h = i % H;
  ev[i] /= fmaxf(den[dstv[e] * H + h], 1e-16f);
}
__global__ void k_attn_agg(const float* __restrict__ al, const float* __restrict__ v,
                           const int* __restrict__ src, const int* __restrict__ dst, float* agg) {
  long i = (long)blockIdx.x * blockDim.x + threadIdx.x;
  if (i >= (long)EE * (HT*EMB)) return;
  int e = (int)(i / (HT*EMB)); int j = (int)(i % (HT*EMB)); int h = j / EMB;
  float a = al[e * HT + h];
  if (a != 0.f) atomicAdd(&agg[(size_t)dst[e] * (HT*EMB) + j], a * v[(size_t)src[e] * (HT*EMB) + j]);
}

// ---------------------------------------------------------------- GAT
__global__ void k_ea_stats(const float* ea, const int* src, const int* dst,
                           const int* em, float* acc) {
  int i = blockIdx.x*blockDim.x+threadIdx.x; if (i >= EE) return;
  if (!em[i] || src[i] == dst[i]) return;
  for (int f = 0; f < EFEAT; ++f) atomicAdd(&acc[f], ea[(size_t)i * EFEAT + f]);
  atomicAdd(&acc[EFEAT], 1.f);
}
__global__ void k_ea_fin(float* acc) {
  int f = threadIdx.x; if (f >= EFEAT) return;
  float c = fmaxf(acc[EFEAT], 1.f);
  acc[f] /= c;
}
// wa[h,f] = sum_c We[f, h*EMB+c] * ae[h,c]   (collapses eattr@We projection)
__global__ void k_we_ae(const float* We, const float* ae, float* wa) {
  int i = threadIdx.x; if (i >= HG * EFEAT) return;
  int h = i / EFEAT, f = i % EFEAT;
  float s = 0.f;
  for (int c = 0; c < EMB; ++c) s += We[(size_t)f * (HG*EMB) + h * EMB + c] * ae[h * EMB + c];
  wa[i] = s;
}
__global__ void k_aedge(const float* ea, const float* wa, float* out) {
  int i = blockIdx.x*blockDim.x+threadIdx.x; if (i >= EE * HG) return;
  int e = i / HG, h = i % HG;
  float s = 0.f;
  for (int f = 0; f < EFEAT; ++f) s += ea[(size_t)e * EFEAT + f] * wa[h * EFEAT + f];
  out[i] = s;
}
__global__ void k_aedge_self(const float* mean, const float* wa, float* out) {
  int h = threadIdx.x; if (h >= HG) return;
  float s = 0.f;
  for (int f = 0; f < EFEAT; ++f) s += mean[f] * wa[h * EFEAT + f];
  out[h] = s;
}
__global__ void k_edges2(const int* src, const int* dst, const int* em, const int* nm,
                         int* s2, int* d2, int* m2) {
  int i = blockIdx.x*blockDim.x+threadIdx.x; if (i >= EE + NN) return;
  if (i < EE) { int s = src[i], d = dst[i]; s2[i] = s; d2[i] = d; m2[i] = (em[i] && (s != d)) ? 1 : 0; }
  else        { int n = i - EE; s2[i] = n; d2[i] = n; m2[i] = nm[n] ? 1 : 0; }
}
__global__ void k_nodescore(const float* __restrict__ xs, const float* __restrict__ as_,
                            const float* __restrict__ ad_, float* asrc, float* adst) {
  int i = blockIdx.x*blockDim.x+threadIdx.x; if (i >= NN * HG) return;
  int n = i / HG, h = i % HG;
  const float* xp = xs + (size_t)n * (HG*EMB) + h * EMB;
  float s = 0.f, d = 0.f;
  for (int c = 0; c < EMB; ++c) { float x = xp[c]; s += x * as_[h*EMB+c]; d += x * ad_[h*EMB+c]; }
  asrc[i] = s; adst[i] = d;
}
__global__ void k_gat_logits(const float* asrc, const float* adst, const float* aedge,
                             const float* aes, const int* s2, const int* d2, float* lg) {
  int i = blockIdx.x*blockDim.x+threadIdx.x; if (i >= (EE + NN) * HG) return;
  int e = i / HG, h = i % HG;
  float ae = (e < EE) ? aedge[e * HG + h] : aes[h];
  float l = asrc[s2[e] * HG + h] + adst[d2[e] * HG + h] + ae;
  lg[i] = (l > 0.f) ? l : 0.2f * l;   // leaky_relu 0.2
}
__global__ void k_gat_scatter(const float* __restrict__ al, const float* __restrict__ xs,
                              const int* s2, const int* d2, const int* m2, float* out) {
  long i = (long)blockIdx.x * blockDim.x + threadIdx.x;
  if (i >= (long)(EE + NN) * (HG*EMB)) return;
  int e = (int)(i / (HG*EMB)); int j = (int)(i % (HG*EMB));
  if (!m2[e]) return;
  int h = j / EMB;
  float a = al[e * HG + h];
  if (a != 0.f) atomicAdd(&out[(size_t)d2[e] * (HG*EMB) + j], a * xs[(size_t)s2[e] * (HG*EMB) + j]);
}
__global__ void k_bias_mask(float* out, const float* b, const int* nm, int C) {
  int i = blockIdx.x*blockDim.x+threadIdx.x; if (i >= NN * C) return;
  int n = i / C, j = i % C;
  out[i] = nm[n] ? (out[i] + b[j]) : 0.f;
}

// ---------------------------------------------------------------- top-k / pool / head
__global__ void k_wnorm(const float* w, float* s) {
  __shared__ float sh[256];
  float v = w[threadIdx.x]; sh[threadIdx.x] = v * v; __syncthreads();
  for (int st = 128; st > 0; st >>= 1) { if (threadIdx.x < st) sh[threadIdx.x] += sh[threadIdx.x + st]; __syncthreads(); }
  if (threadIdx.x == 0) s[0] = sqrtf(sh[0]);
}
__global__ void k_score(const float* x, const float* w, const float* wn, float* sc) {
  int n = blockIdx.x*blockDim.x+threadIdx.x; if (n >= NN) return;
  float s = 0.f;
  for (int c = 0; c < EMB; ++c) s += x[(size_t)n * EMB + c] * w[c];
  sc[n] = s / wn[0];
}
__global__ void k_topk(const float* sc, int* nm, const float* xin, float* xout, int kk) {
  __shared__ float s[NPER]; __shared__ int act[NPER];
  int g = blockIdx.x, t = threadIdx.x, n = g * NPER + t;
  act[t] = nm[n];
  s[t] = act[t] ? sc[n] : -__builtin_inff();
  __syncthreads();
  int cnt = 0; float mys = s[t];
  for (int m = 0; m < NPER; ++m)
    if (act[m] && (s[m] > mys || (s[m] == mys && m < t))) cnt++;
  int keep = act[t] && (cnt < kk);
  float factor = keep ? tanhf(sc[n]) : 0.f;
  for (int c = 0; c < EMB; ++c) xout[(size_t)n * EMB + c] = xin[(size_t)n * EMB + c] * factor;
  nm[n] = keep;
}
__global__ void k_emask_upd(int* em, const int* src, const int* dst, const int* nm) {
  int i = blockIdx.x*blockDim.x+threadIdx.x; if (i >= EE) return;
  em[i] = (em[i] && nm[src[i]] && nm[dst[i]]) ? 1 : 0;
}
__global__ void k_gap(const float* x, const int* nm, float* out) {
  int g = blockIdx.x, c = threadIdx.x;
  float s = 0.f, cnt = 0.f;
  for (int i = 0; i < NPER; ++i) {
    int n = g * NPER + i;
    if (nm[n]) { s += x[(size_t)n * EMB + c]; cnt += 1.f; }
  }
  out[(size_t)g * EMB + c] = s / fmaxf(cnt, 1.f);
}
__global__ void k_add3(const float* a, const float* b, const float* c, float* o, int n) {
  int i = blockIdx.x*blockDim.x+threadIdx.x; if (i < n) o[i] = a[i] + b[i] + c[i];
}
__global__ void k_lin2(const float* h, const float* w, const float* b, float* out) {
  int g = blockIdx.x*blockDim.x+threadIdx.x; if (g >= G_) return;
  float s = 0.f;
  for (int j = 0; j < 1024; ++j) s += h[(size_t)g * 1024 + j] * w[j];
  out[g] = s + b[0];
}

// ---------------------------------------------------------------- host
static inline int cdiv(int a, int b) { return (a + b - 1) / b; }
static inline unsigned fenc_host(float f) {
  unsigned u; memcpy(&u, &f, 4);
  return (u & 0x80000000u) ? ~u : (u | 0x80000000u);
}

extern "C" void kernel_launch(void* const* d_in, const int* in_sizes, int n_in,
                              void* d_out, int out_size, void* d_ws, size_t ws_size,
                              hipStream_t stream) {
  (void)in_sizes; (void)n_in; (void)out_size; (void)ws_size;
  const float* x_in  = (const float*)d_in[0];
  const float* ea_in = (const float*)d_in[1];
  const int*   ei    = (const int*)d_in[2];
  const int*   src   = ei;
  const int*   dst   = ei + EE;
#define P(i) ((const float*)d_in[i])
  // params: 4 gin_W1,5 gin_b1,6 gin_W2,7 gin_b2, 8..15 tr_Wq/bq/Wk/bk/Wv/bv/Wskip/bskip,
  // 16 tr_red_W,17 tr_red_b, 18 lin1_W,19 lin1_b, 20 lin2_W,21 lin2_b,
  // then per layer i (base 22 + 9*i): W,as,ad,We,ae,b,headW,headb,poolw

  char* base = (char*)d_ws; size_t off = 0;
  auto alloc = [&](size_t bytes) -> void* { void* p = base + off; off = (off + bytes + 255) & ~(size_t)255; return p; };

  bf16_t* ginW1T = (bf16_t*)alloc((size_t)256 * 32 * 2);
  bf16_t* ginW2T = (bf16_t*)alloc((size_t)256 * 256 * 2);
  bf16_t* WqT    = (bf16_t*)alloc((size_t)1024 * 256 * 2);
  bf16_t* WkT    = (bf16_t*)alloc((size_t)1024 * 256 * 2);
  bf16_t* WvT    = (bf16_t*)alloc((size_t)1024 * 256 * 2);
  bf16_t* WskT   = (bf16_t*)alloc((size_t)1024 * 256 * 2);
  bf16_t* redT   = (bf16_t*)alloc((size_t)256 * 1024 * 2);
  bf16_t* lin1T  = (bf16_t*)alloc((size_t)1024 * 256 * 2);
  bf16_t* gatWT[3], *headWT[3];
  for (int i = 0; i < 3; ++i) {
    gatWT[i]  = (bf16_t*)alloc((size_t)768 * 256 * 2);
    headWT[i] = (bf16_t*)alloc((size_t)256 * 768 * 2);
  }
  bf16_t* hb   = (bf16_t*)alloc((size_t)NN * 32 * 2);
  bf16_t* BF   = (bf16_t*)alloc((size_t)NN * 1024 * 2);
  float*  FA   = (float*)alloc((size_t)NN * 1024 * 4);
  float*  FB   = (float*)alloc((size_t)NN * 1024 * 4);
  float*  X    = (float*)alloc((size_t)NN * EMB * 4);
  float*  X2   = (float*)alloc((size_t)NN * EMB * 4);
  float*  lg   = (float*)alloc((size_t)(EE + NN) * HT * 4);
  float*  ev   = (float*)alloc((size_t)(EE + NN) * HT * 4);
  unsigned* segm = (unsigned*)alloc((size_t)NN * HT * 4);
  float*  segden = (float*)alloc((size_t)NN * HT * 4);
  float*  asrc = (float*)alloc((size_t)NN * HG * 4);
  float*  adst = (float*)alloc((size_t)NN * HG * 4);
  float*  aedge = (float*)alloc((size_t)EE * HG * 4);
  float*  smallb = (float*)alloc(64 * 4);   // [0..10]=ea_mean [11]=cnt [16..48]=we_ae [52..54]=aedge_self [56]=wnorm
  int* s2 = (int*)alloc((size_t)(EE + NN) * 4);
  int* d2 = (int*)alloc((size_t)(EE + NN) * 4);
  int* m2 = (int*)alloc((size_t)(EE + NN) * 4);
  int* nmask = (int*)alloc((size_t)NN * 4);
  int* emask = (int*)alloc((size_t)EE * 4);
  float* score = (float*)alloc((size_t)NN * 4);
  float* outs  = (float*)alloc((size_t)3 * G_ * EMB * 4);
  float* xsum  = (float*)alloc((size_t)G_ * EMB * 4);
  float* l1o   = (float*)alloc((size_t)G_ * 1024 * 4);

#define L1(kern, n, ...) kern<<<cdiv((n), 256), 256, 0, stream>>>(__VA_ARGS__)
  auto gemm = [&](const bf16_t* A, const bf16_t* BT, float* C, const float* bias,
                  int M, int Nc, int K, int relu, int accum) {
    dim3 grid(cdiv(Nc, 16 * TN * GEMM_WAVES), cdiv(M, 16 * TM));
    k_gemm_bf16<<<grid, 32 * GEMM_WAVES, 0, stream>>>(A, BT, C, bias, M, Nc, K, relu, accum);
  };

  // ---- weight prep (bf16, transposed, K padded to mult of 32)
  L1(k_wt, 256 * 32,    P(4),  ginW1T, FEATN, 256, 32);
  L1(k_wt, 256 * 256,   P(6),  ginW2T, 256, 256, 256);
  L1(k_wt, 1024 * 256,  P(8),  WqT,  256, 1024, 256);
  L1(k_wt, 1024 * 256,  P(10), WkT,  256, 1024, 256);
  L1(k_wt, 1024 * 256,  P(12), WvT,  256, 1024, 256);
  L1(k_wt, 1024 * 256,  P(14), WskT, 256, 1024, 256);
  L1(k_wt, 256 * 1024,  P(16), redT, 1024, 256, 1024);
  L1(k_wt, 1024 * 256,  P(18), lin1T, 256, 1024, 256);
  for (int i = 0; i < 3; ++i) {
    L1(k_wt, 768 * 256, P(22 + 9 * i),     gatWT[i],  256, 768, 256);
    L1(k_wt, 256 * 768, P(22 + 9 * i + 6), headWT[i], 768, 256, 768);
  }
  L1(k_fill_int, NN, nmask, 1, NN);
  L1(k_fill_int, EE, emask, 1, EE);

  // ---- GIN: h = x + segsum(x[src] -> dst); x = relu(h@W1+b1)@W2+b2
  L1(k_copy_f32, NN * FEATN, x_in, FB, NN * FEATN);
  L1(k_gin_scatter, EE * FEATN, x_in, src, dst, FB);
  L1(k_cvt_pad, NN * 32, FB, hb, NN, FEATN, 32);
  gemm(hb, ginW1T, FA, P(5), NN, 256, 32, 1, 0);
  L1(k_cvt_bf16, NN * 256, FA, BF, NN * 256);
  gemm(BF, ginW2T, X, P(7), NN, 256, 256, 0, 0);

  // ---- TransformerConv
  L1(k_cvt_bf16, NN * 256, X, BF, NN * 256);               // xb
  gemm(BF, WqT, FA, P(9),  NN, 1024, 256, 0, 0);           // q
  gemm(BF, WkT, FB, P(11), NN, 1024, 256, 0, 0);           // k
  L1(k_attn_logits, EE * HT, FA, FB, src, dst, lg);
  L1(k_fill_u32, NN * HT, segm, fenc_host(-1e30f), NN * HT);
  L1(k_zero_f32, NN * HT, segden, NN * HT);
  L1(k_seg_max, EE * HT, lg, dst, (const int*)nullptr, segm, EE, HT);
  L1(k_seg_exp, EE * HT, lg, dst, (const int*)nullptr, segm, ev, segden, EE, HT);
  L1(k_seg_div, EE * HT, ev, segden, dst, EE, HT);
  gemm(BF, WvT, FA, P(13), NN, 1024, 256, 0, 0);           // v (q dead)
  L1(k_zero_f32, NN * 1024, FB, NN * 1024);                // agg (k dead)
  k_attn_agg<<<cdiv(EE * (HT * EMB), 256), 256, 0, stream>>>(ev, FA, src, dst, FB);
  gemm(BF, WskT, FB, P(15), NN, 1024, 256, 0, 1);          // FB += x@Wskip + bskip
  L1(k_cvt_bf16, NN * 1024, FB, BF, NN * 1024);
  gemm(BF, redT, X, P(17), NN, 256, 1024, 0, 0);           // x = concat @ red + b

  // ---- 3x (GAT -> head -> topk -> gap)
  const int kvals[3] = {52, 26, 8};
  for (int i = 0; i < 3; ++i) {
    const int pb = 22 + 9 * i;
    float* ea_mean = smallb;           // [0..10], cnt at [11]
    float* we_ae   = smallb + 16;      // [3*11]
    float* ae_self = smallb + 52;      // [3]
    float* wnorm   = smallb + 56;

    L1(k_zero_f32, 64, smallb, 64);
    L1(k_ea_stats, EE, ea_in, src, dst, emask, ea_mean);
    k_ea_fin<<<1, 32, 0, stream>>>(ea_mean);
    k_we_ae<<<1, 64, 0, stream>>>(P(pb + 3), P(pb + 4), we_ae);
    L1(k_aedge, EE * HG, ea_in, we_ae, aedge);
    k_aedge_self<<<1, 32, 0, stream>>>(ea_mean, we_ae, ae_self);

    L1(k_cvt_bf16, NN * 256, X, BF, NN * 256);
    gemm(BF, gatWT[i], FA, nullptr, NN, 768, 256, 0, 0);   // xs = x @ W
    L1(k_nodescore, NN * HG, FA, P(pb + 1), P(pb + 2), asrc, adst);
    L1(k_edges2, EE + NN, src, dst, emask, nmask, s2, d2, m2);
    L1(k_gat_logits, (EE + NN) * HG, asrc, adst, aedge, ae_self, s2, d2, lg);
    L1(k_fill_u32, NN * HG, segm, fenc_host(-1e30f), NN * HG);
    L1(k_zero_f32, NN * HG, segden, NN * HG);
    L1(k_seg_max, (EE + NN) * HG, lg, d2, m2, segm, EE + NN, HG);
    L1(k_seg_exp, (EE + NN) * HG, lg, d2, m2, segm, ev, segden, EE + NN, HG);
    L1(k_seg_div, (EE + NN) * HG, ev, segden, d2, EE + NN, HG);
    L1(k_zero_f32, NN * 768, FB, NN * 768);
    k_gat_scatter<<<cdiv((EE + NN) * (HG * EMB), 256), 256, 0, stream>>>(ev, FA, s2, d2, m2, FB);
    L1(k_bias_mask, NN * 768, FB, P(pb + 5), nmask, 768);
    L1(k_cvt_bf16, NN * 768, FB, BF, NN * 768);
    gemm(BF, headWT[i], X2, P(pb + 7), NN, 256, 768, 0, 0);

    k_wnorm<<<1, 256, 0, stream>>>(P(pb + 8), wnorm);
    L1(k_score, NN, X2, P(pb + 8), wnorm, score);
    k_topk<<<G_, NPER, 0, stream>>>(score, nmask, X2, X, kvals[i]);
    L1(k_emask_upd, EE, emask, src, dst, nmask);
    k_gap<<<G_, EMB, 0, stream>>>(X, nmask, outs + (size_t)i * G_ * EMB);
  }

  // ---- head MLP
  L1(k_add3, G_ * EMB, outs, outs + (size_t)G_ * EMB, outs + (size_t)2 * G_ * EMB, xsum, G_ * EMB);
  L1(k_cvt_bf16, G_ * EMB, xsum, BF, G_ * EMB);
  gemm(BF, lin1T, l1o, P(19), G_, 1024, 256, 1, 0);
  k_lin2<<<1, 128, 0, stream>>>(l1o, P(20), P(21), (float*)d_out);
}
